// MixTransformerBlock_34651796144666
// MI455X (gfx1250) — matrix operand order repaired and checked by reading, not compiled, not run on hardware
//
#include <hip/hip_runtime.h>
#include <hip/hip_bf16.h>
#include <math.h>

// ---------------------------------------------------------------------------
// MixTransformerBlock for MI455X (gfx1250, wave32, WMMA).
// All GEMMs use v_wmma_f32_16x16x32_bf16 (bf16 inputs, f32 accumulate).
// LayerNorm / softmax / GELU / residuals in f32.
// GEMM K-loop is software-pipelined: double-buffered LDS tiles, A staged with
// GLOBAL_LOAD_ASYNC_TO_LDS_B128 (ASYNCcnt) issued for tile i+1 before the
// WMMAs of tile i, drained with s_wait_asynccnt at iteration end.
// Requires ws_size >= ~400 MB (buffers reused across phases).
// ---------------------------------------------------------------------------

typedef __bf16 bf16_t;
typedef __attribute__((ext_vector_type(16))) __bf16 v16bf;
typedef __attribute__((ext_vector_type(8)))  float  v8f;

#define ROWS   32768   // B * H * W = 8 * 64 * 64
#define CDIM   512
#define NHEAD  16
#define HDIM   32
#define NWIN   512     // B * (64/8) * (64/8)
#define NTOK   64      // 8*8 tokens per window
#define MLPH   2048

union FragU { v16bf v; uint4 u[2]; bf16_t e[16]; };

// --- async global -> LDS copy (gfx1250, tracked with ASYNCcnt) --------------
// GV addressing: vdst = per-lane LDS byte address, vaddr = 64-bit global addr.
__device__ inline void async_copy_b128(void* lds_dst, const void* gsrc) {
  const unsigned lds_addr = (unsigned)(uintptr_t)lds_dst;        // addr[31:0] = LDS offset
  const unsigned long long ga = (unsigned long long)(uintptr_t)gsrc;
  asm volatile("global_load_async_to_lds_b128 %0, %1, off"
               :: "v"(lds_addr), "v"(ga)
               : "memory");
}
__device__ inline void wait_async() {
  asm volatile("s_wait_asynccnt 0x0" ::: "memory");
}

// --- WMMA fragment loaders (wave32; contiguous k order, same in A and B) -----------
// A (16x32 bf16): lane m=L&15, half=L>>4; elems 0..15 = K half*16..half*16+15
//                 (contiguous)  -> two 16B loads.
__device__ inline v16bf lds_frag_a(const bf16_t* base, int ld, int lane) {
  const int m = lane & 15, half = lane >> 4;
  FragU r;
  r.u[0] = *reinterpret_cast<const uint4*>(base + m * ld + half * 16);
  r.u[1] = *reinterpret_cast<const uint4*>(base + m * ld + half * 16 + 8);
  return r.v;
}
// B (32x16 bf16) loaded from an N-major (transposed) LDS tile Bt[n][k]:
// lane n=L&15, K = (L>>4)*16 + e  -> 16 contiguous bf16 = two 16B loads.
__device__ inline v16bf lds_frag_b(const bf16_t* base, int ld, int k0, int lane) {
  const int n = lane & 15, half = lane >> 4;
  const bf16_t* p = base + n * ld + k0 + half * 16;
  FragU r;
  r.u[0] = *reinterpret_cast<const uint4*>(p);
  r.u[1] = *reinterpret_cast<const uint4*>(p + 8);
  return r.v;
}

__device__ inline v8f wmma_bf16(v16bf a, v16bf b, v8f c) {
  return __builtin_amdgcn_wmma_f32_16x16x32_bf16(false, a, false, b, (short)0, c,
                                                 false, false);
}

__device__ inline v8f v8f_zero() {
  v8f z = {0.f, 0.f, 0.f, 0.f, 0.f, 0.f, 0.f, 0.f};
  return z;
}

__device__ inline float gelu_exact(float x) {
  return 0.5f * x * (1.0f + erff(x * 0.70710678118654752f));
}

// window index (0..511), token t (0..63) -> flat row in (B, H*W)
__device__ inline int win_row(int wi, int t) {
  const int b  = wi >> 6, w2 = wi & 63;
  const int wh = w2 >> 3, ww = w2 & 7;
  const int i  = t >> 3,  j  = t & 7;
  return b * 4096 + (wh * 8 + i) * 64 + (ww * 8 + j);
}

// --- elementwise f32 -> bf16 cast ------------------------------------------
__global__ __launch_bounds__(256)
void cast_f32_bf16_kernel(const float* __restrict__ src, bf16_t* __restrict__ dst, int n) {
  const int i = blockIdx.x * 256 + threadIdx.x;
  if (i < n) dst[i] = (bf16_t)src[i];
}

// --- LayerNorm over last dim (512), f32 in -> bf16 out ---------------------
__global__ __launch_bounds__(128)
void ln_kernel(const float* __restrict__ x, const float* __restrict__ g,
               const float* __restrict__ b, bf16_t* __restrict__ out) {
  __shared__ float red[128];
  __shared__ float red2[128];
  const int row = blockIdx.x;
  const int tid = threadIdx.x;
  const float4 v = *reinterpret_cast<const float4*>(x + (size_t)row * CDIM + tid * 4);
  float vv[4] = {v.x, v.y, v.z, v.w};
  float s = vv[0] + vv[1] + vv[2] + vv[3];
  float s2 = vv[0]*vv[0] + vv[1]*vv[1] + vv[2]*vv[2] + vv[3]*vv[3];
  red[tid] = s; red2[tid] = s2;
  __syncthreads();
  for (int off = 64; off > 0; off >>= 1) {
    if (tid < off) { red[tid] += red[tid + off]; red2[tid] += red2[tid + off]; }
    __syncthreads();
  }
  const float mean = red[0] * (1.0f / CDIM);
  const float var  = red2[0] * (1.0f / CDIM) - mean * mean;
  const float rs   = rsqrtf(var + 1e-4f);
  #pragma unroll
  for (int i = 0; i < 4; ++i) {
    const int c = tid * 4 + i;
    out[(size_t)row * CDIM + c] = (bf16_t)((vv[i] - mean) * rs * g[c] + b[c]);
  }
}

// --- Tiled WMMA GEMM: Out[M,N] = epilogue(A[M,K] @ W[K,N] + bias) ----------
// Block tile 128x128, 8 waves (4x2), each wave 32x64 = 8 wmma per K-step of 32.
// Double-buffered LDS; A-tile via async global->LDS overlapped with compute.
// EPI: 0 bias->bf16, 1 (bias)*scale->bf16, 2 gelu(bias)->bf16,
//      3 resid + bias -> f32
template<int EPI>
__global__ __launch_bounds__(256)
void gemm_bf16_kernel(const bf16_t* __restrict__ A, const bf16_t* __restrict__ W,
                      const float* __restrict__ bias, const float* __restrict__ resid,
                      bf16_t* __restrict__ outb, float* __restrict__ outf,
                      int M, int N, int K, float scale) {
  __shared__ bf16_t As[2][128 * 40];  // [m][k], stride 40 (80B rows, 16B aligned)
  __shared__ bf16_t Bs[2][128 * 40];  // transposed: [n][k]
  const int tid  = threadIdx.x;
  const int lane = tid & 31, wid = tid >> 5;
  const int wm = wid & 3, wn = wid >> 2;          // wave tile: 32(M) x 64(N)
  const int m0 = blockIdx.y * 128, n0 = blockIdx.x * 128;

  v8f acc[8];                                     // [mi(2)][ni(4)]
  #pragma unroll
  for (int i = 0; i < 8; ++i) acc[i] = v8f_zero();

  const int ar = tid >> 1;            // A: 0..127 rows
  const int ac = (tid & 1) * 16;      // A: col 0 / 16
  const int bk = tid >> 3;            // B: 0..31 k-rows
  const int bn = (tid & 7) * 16;      // B: 16-wide n segment

  // ---- prologue: stage tile 0 -------------------------------------------
  {
    const bf16_t* ga = A + (size_t)(m0 + ar) * K + ac;
    async_copy_b128(&As[0][ar * 40 + ac],     ga);
    async_copy_b128(&As[0][ar * 40 + ac + 8], ga + 8);
    FragU t;
    const bf16_t* gb = W + (size_t)bk * N + n0 + bn;
    t.u[0] = *reinterpret_cast<const uint4*>(gb);
    t.u[1] = *reinterpret_cast<const uint4*>(gb + 8);
    #pragma unroll
    for (int i = 0; i < 16; ++i) Bs[0][(bn + i) * 40 + bk] = t.e[i];
  }
  wait_async();
  __syncthreads();

  // ---- pipelined K loop --------------------------------------------------
  #pragma unroll 2
  for (int kk = 0; kk < K; kk += 32) {
    const int cur = (kk >> 5) & 1, nxt = cur ^ 1;
    const bool has_next = (kk + 32) < K;
    FragU breg;
    if (has_next) {  // issue next tile's transfers before computing
      const bf16_t* ga = A + (size_t)(m0 + ar) * K + kk + 32 + ac;
      async_copy_b128(&As[nxt][ar * 40 + ac],     ga);
      async_copy_b128(&As[nxt][ar * 40 + ac + 8], ga + 8);
      const bf16_t* gb = W + (size_t)(kk + 32 + bk) * N + n0 + bn;
      breg.u[0] = *reinterpret_cast<const uint4*>(gb);
      breg.u[1] = *reinterpret_cast<const uint4*>(gb + 8);
      __builtin_prefetch(gb + 32 * N, 0, 1);     // global_prefetch_b8
    }
    // compute on current tile: 6 fragment loads -> 8 wmma
    v16bf a0 = lds_frag_a(&As[cur][(wm * 32)      * 40], 40, lane);
    v16bf a1 = lds_frag_a(&As[cur][(wm * 32 + 16) * 40], 40, lane);
    #pragma unroll
    for (int ni = 0; ni < 4; ++ni) {
      v16bf b = lds_frag_b(&Bs[cur][(wn * 64 + ni * 16) * 40], 40, 0, lane);
      acc[ni]     = wmma_bf16(a0, b, acc[ni]);
      acc[4 + ni] = wmma_bf16(a1, b, acc[4 + ni]);
    }
    if (has_next) {  // finish staging next tile, then converge
      #pragma unroll
      for (int i = 0; i < 16; ++i) Bs[nxt][(bn + i) * 40 + bk] = breg.e[i];
      wait_async();
    }
    __syncthreads();
  }

  // ---- epilogue ----------------------------------------------------------
  const int half = lane >> 4, nl = lane & 15;
  #pragma unroll
  for (int mi = 0; mi < 2; ++mi) {
    #pragma unroll
    for (int ni = 0; ni < 4; ++ni) {
      const int gn = n0 + wn * 64 + ni * 16 + nl;
      const float bv = bias[gn];
      #pragma unroll
      for (int r = 0; r < 8; ++r) {
        const int gm = m0 + wm * 32 + mi * 16 + r + half * 8;
        const size_t o = (size_t)gm * N + gn;
        float v = acc[mi * 4 + ni][r] + bv;
        if (EPI == 0)      outb[o] = (bf16_t)v;
        else if (EPI == 1) outb[o] = (bf16_t)(v * scale);
        else if (EPI == 2) outb[o] = (bf16_t)gelu_exact(v);
        else               outf[o] = resid[o] + v;
      }
    }
  }
}

// --- Windowed cross-attention: one block per (window, head) ----------------
// Q pre-scaled bf16 [ROWS,512]; KV bf16 [ROWS,1024] (k | v, head*32 slices).
// O bf16 [ROWS,512] in token layout (window-reverse folded into the scatter).
__global__ __launch_bounds__(128)
void attn_kernel(const bf16_t* __restrict__ Q, const bf16_t* __restrict__ KV,
                 const float* __restrict__ nuclei, const float* __restrict__ rpb,
                 bf16_t* __restrict__ O) {
  __shared__ bf16_t Qs[64 * 40];   // [q][d]
  __shared__ bf16_t Ks[64 * 40];   // [k_tok][d]  == Bt layout for S = Q*K^T
  __shared__ bf16_t Vt[32 * 72];   // [d][k_tok]  == Bt layout for O = P*V
  __shared__ bf16_t Pb[64 * 72];   // probabilities (A-matrix for 2nd matmul)
  __shared__ float  Sf[64 * 72];   // scores f32
  __shared__ float  nmw[64];       // nuclei mask per window token

  const int tid = threadIdx.x;
  const int lane = tid & 31, wid = tid >> 5;
  const int wi = blockIdx.x >> 4, head = blockIdx.x & 15;

  { // cooperative loads: 2 threads per token, 16 channels each
    const int t = tid >> 1, d0 = (tid & 1) * 16;
    const int row = win_row(wi, t);
    // Q and K tiles are pure copies -> async global->LDS
    const bf16_t* gq = Q + (size_t)row * 512 + head * 32 + d0;
    async_copy_b128(&Qs[t * 40 + d0],     gq);
    async_copy_b128(&Qs[t * 40 + d0 + 8], gq + 8);
    const bf16_t* gk = KV + (size_t)row * 1024 + head * 32 + d0;
    async_copy_b128(&Ks[t * 40 + d0],     gk);
    async_copy_b128(&Ks[t * 40 + d0 + 8], gk + 8);
    // V tile needs a transpose -> stage through VGPRs
    FragU tv;
    const bf16_t* gv = KV + (size_t)row * 1024 + 512 + head * 32 + d0;
    tv.u[0] = *reinterpret_cast<const uint4*>(gv);
    tv.u[1] = *reinterpret_cast<const uint4*>(gv + 8);
    #pragma unroll
    for (int i = 0; i < 16; ++i) Vt[(d0 + i) * 72 + t] = tv.e[i];
    if ((tid & 1) == 0) nmw[t] = nuclei[row];
  }
  wait_async();
  __syncthreads();

  { // S = Q @ K^T  (+ rel-pos bias + nuclei mask); K=HD=32 -> 1 wmma per tile
    const int m0 = wid * 16;
    const int half = lane >> 4, nl = lane & 15;
    v16bf aq = lds_frag_a(&Qs[m0 * 40], 40, lane);
    #pragma unroll
    for (int nt = 0; nt < 4; ++nt) {
      const int n0 = nt * 16;
      v16bf bk = lds_frag_b(&Ks[n0 * 40], 40, 0, lane);
      v8f s = wmma_bf16(aq, bk, v8f_zero());
      #pragma unroll
      for (int r = 0; r < 8; ++r) {
        const int m = m0 + r + half * 8, n = n0 + nl;
        const int idx = ((m >> 3) - (n >> 3) + 7) * 15 + ((m & 7) - (n & 7) + 7);
        Sf[m * 72 + n] = s[r] + rpb[idx * NHEAD + head] + nmw[m] + nmw[n];
      }
    }
  }
  __syncthreads();

  if (tid < 64) { // row softmax in f32, result as bf16 in Pb
    float mx = -1e30f;
    for (int n = 0; n < 64; ++n) mx = fmaxf(mx, Sf[tid * 72 + n]);
    float sum = 0.f;
    for (int n = 0; n < 64; ++n) {
      const float e = expf(Sf[tid * 72 + n] - mx);
      Sf[tid * 72 + n] = e;
      sum += e;
    }
    const float inv = 1.0f / sum;
    for (int n = 0; n < 64; ++n) Pb[tid * 72 + n] = (bf16_t)(Sf[tid * 72 + n] * inv);
  }
  __syncthreads();

  { // O = P @ V  (K=64 -> 2 wmma per 16x16 tile)
    const int m0 = wid * 16;
    const int half = lane >> 4, nl = lane & 15;
    #pragma unroll
    for (int nt = 0; nt < 2; ++nt) {
      const int n0 = nt * 16;
      v8f acc = v8f_zero();
      #pragma unroll
      for (int k0 = 0; k0 < 64; k0 += 32) {
        v16bf ap = lds_frag_a(&Pb[m0 * 72 + k0], 72, lane);
        v16bf bv = lds_frag_b(&Vt[n0 * 72], 72, k0, lane);
        acc = wmma_bf16(ap, bv, acc);
      }
      #pragma unroll
      for (int r = 0; r < 8; ++r) {
        const int m = m0 + r + half * 8;
        O[(size_t)win_row(wi, m) * 512 + head * 32 + n0 + nl] = (bf16_t)acc[r];
      }
    }
  }
}

// ---------------------------------------------------------------------------
extern "C" void kernel_launch(void* const* d_in, const int* in_sizes, int n_in,
                              void* d_out, int out_size, void* d_ws, size_t ws_size,
                              hipStream_t stream) {
  (void)in_sizes; (void)n_in; (void)out_size; (void)ws_size;
  const float* x      = (const float*)d_in[0];
  const float* y      = (const float*)d_in[1];
  const float* nuc    = (const float*)d_in[2];
  const float* rpb    = (const float*)d_in[3];
  const float* g1     = (const float*)d_in[4];  const float* b1 = (const float*)d_in[5];
  const float* g3     = (const float*)d_in[6];  const float* b3 = (const float*)d_in[7];
  const float* w_qkv  = (const float*)d_in[8];  const float* b_qkv   = (const float*)d_in[9];
  const float* w_qkvy = (const float*)d_in[10]; const float* b_qkv_y = (const float*)d_in[11];
  const float* w_qx   = (const float*)d_in[12]; const float* b_qx    = (const float*)d_in[13];
  const float* w_qy   = (const float*)d_in[14]; const float* b_qy    = (const float*)d_in[15];
  const float* w_proj = (const float*)d_in[16]; const float* b_proj  = (const float*)d_in[17];
  const float* w_projy= (const float*)d_in[18]; const float* b_projy = (const float*)d_in[19];
  const float* g2     = (const float*)d_in[20]; const float* b2 = (const float*)d_in[21];
  const float* g4     = (const float*)d_in[22]; const float* b4 = (const float*)d_in[23];
  const float* w_fc1  = (const float*)d_in[24]; const float* b_fc1  = (const float*)d_in[25];
  const float* w_fc2  = (const float*)d_in[26]; const float* b_fc2  = (const float*)d_in[27];
  const float* w_fc1y = (const float*)d_in[28]; const float* b_fc1y = (const float*)d_in[29];
  const float* w_fc2y = (const float*)d_in[30]; const float* b_fc2y = (const float*)d_in[31];

  char* ws = (char*)d_ws;
  size_t off = 0;
  auto alloc = [&](size_t bytes) -> char* {
    char* p = ws + off;
    off += (bytes + 255) & ~(size_t)255;
    return p;
  };

  // activation scratch (reused across phases)
  bf16_t* lnx = (bf16_t*)alloc((size_t)ROWS * 512 * 2);  // -> attn out x
  bf16_t* lny = (bf16_t*)alloc((size_t)ROWS * 512 * 2);  // -> attn out y
  bf16_t* qx  = (bf16_t*)alloc((size_t)ROWS * 512 * 2);  // -> ln2(x)
  bf16_t* qy  = (bf16_t*)alloc((size_t)ROWS * 512 * 2);  // -> ln4(y)
  float*  x1  = (float*) alloc((size_t)ROWS * 512 * 4);
  float*  y1  = (float*) alloc((size_t)ROWS * 512 * 4);
  bf16_t* big = (bf16_t*)alloc((size_t)ROWS * 2048 * 2); // kvx|kvy, then MLP hidden
  bf16_t* kvx = big;
  bf16_t* kvy = big + (size_t)ROWS * 1024;
  bf16_t* hid = big;
  // bf16 weights
  bf16_t* wqkv  = (bf16_t*)alloc((size_t)512 * 1024 * 2);
  bf16_t* wqkvy = (bf16_t*)alloc((size_t)512 * 1024 * 2);
  bf16_t* wqxb  = (bf16_t*)alloc((size_t)512 * 512 * 2);
  bf16_t* wqyb  = (bf16_t*)alloc((size_t)512 * 512 * 2);
  bf16_t* wprj  = (bf16_t*)alloc((size_t)512 * 512 * 2);
  bf16_t* wprjy = (bf16_t*)alloc((size_t)512 * 512 * 2);
  bf16_t* wfc1  = (bf16_t*)alloc((size_t)512 * 2048 * 2);
  bf16_t* wfc2  = (bf16_t*)alloc((size_t)2048 * 512 * 2);
  bf16_t* wfc1y = (bf16_t*)alloc((size_t)512 * 2048 * 2);
  bf16_t* wfc2y = (bf16_t*)alloc((size_t)2048 * 512 * 2);

  auto cast = [&](const float* s, bf16_t* d, int n) {
    cast_f32_bf16_kernel<<<(n + 255) / 256, 256, 0, stream>>>(s, d, n);
  };
  auto gemm = [&](int epi, const bf16_t* A, const bf16_t* Wt, const float* bias,
                  const float* resid, bf16_t* ob, float* of,
                  int M, int N, int K, float sc) {
    dim3 grid(N / 128, M / 128);
    switch (epi) {
      case 0: gemm_bf16_kernel<0><<<grid, 256, 0, stream>>>(A, Wt, bias, resid, ob, of, M, N, K, sc); break;
      case 1: gemm_bf16_kernel<1><<<grid, 256, 0, stream>>>(A, Wt, bias, resid, ob, of, M, N, K, sc); break;
      case 2: gemm_bf16_kernel<2><<<grid, 256, 0, stream>>>(A, Wt, bias, resid, ob, of, M, N, K, sc); break;
      default: gemm_bf16_kernel<3><<<grid, 256, 0, stream>>>(A, Wt, bias, resid, ob, of, M, N, K, sc); break;
    }
  };

  const float SCALE = 0.17677669529663689f;  // HD^-0.5

  // 1) weights -> bf16
  cast(w_qkv,  wqkv,  512 * 1024);
  cast(w_qkvy, wqkvy, 512 * 1024);
  cast(w_qx,   wqxb,  512 * 512);
  cast(w_qy,   wqyb,  512 * 512);
  cast(w_proj, wprj,  512 * 512);
  cast(w_projy,wprjy, 512 * 512);
  cast(w_fc1,  wfc1,  512 * 2048);
  cast(w_fc2,  wfc2,  2048 * 512);
  cast(w_fc1y, wfc1y, 512 * 2048);
  cast(w_fc2y, wfc2y, 2048 * 512);

  // 2) pre-attention LayerNorms
  ln_kernel<<<ROWS, 128, 0, stream>>>(x, g1, b1, lnx);
  ln_kernel<<<ROWS, 128, 0, stream>>>(y, g3, b3, lny);

  // 3) projections (window partition commutes with row-wise GEMM)
  gemm(0, lnx, wqkv,  b_qkv,   nullptr, kvx, nullptr, ROWS, 1024, 512, 1.f);
  gemm(0, lny, wqkvy, b_qkv_y, nullptr, kvy, nullptr, ROWS, 1024, 512, 1.f);
  gemm(1, lnx, wqxb,  b_qx,    nullptr, qx,  nullptr, ROWS, 512,  512, SCALE);
  gemm(1, lny, wqyb,  b_qy,    nullptr, qy,  nullptr, ROWS, 512,  512, SCALE);

  // 4) cross attention: attn_x = softmax(q_y k_x^T + bias + m) v_x -> ox (lnx)
  //                     attn_y = softmax(q_x k_y^T + bias + m) v_y -> oy (lny)
  attn_kernel<<<NWIN * NHEAD, 128, 0, stream>>>(qy, kvx, nuc, rpb, lnx);
  attn_kernel<<<NWIN * NHEAD, 128, 0, stream>>>(qx, kvy, nuc, rpb, lny);

  // 5) output projection + residual -> f32
  gemm(3, lnx, wprj,  b_proj,  x, nullptr, x1, ROWS, 512, 512, 1.f);
  gemm(3, lny, wprjy, b_projy, y, nullptr, y1, ROWS, 512, 512, 1.f);

  // 6) MLP layernorms
  ln_kernel<<<ROWS, 128, 0, stream>>>(x1, g2, b2, qx);
  ln_kernel<<<ROWS, 128, 0, stream>>>(y1, g4, b4, qy);

  // 7) MLPs (hidden buffer reused between streams; stream order serializes)
  float* out_x = (float*)d_out;
  float* out_y = out_x + (size_t)ROWS * 512;
  gemm(2, qx,  wfc1,  b_fc1,  nullptr, hid, nullptr, ROWS, 2048, 512,  1.f);
  gemm(3, hid, wfc2,  b_fc2,  x1,      nullptr, out_x, ROWS, 512, 2048, 1.f);
  gemm(2, qy,  wfc1y, b_fc1y, nullptr, hid, nullptr, ROWS, 2048, 512,  1.f);
  gemm(3, hid, wfc2y, b_fc2y, y1,      nullptr, out_y, ROWS, 512, 2048, 1.f);
}
